// GNN_GIN_60868276519663
// MI455X (gfx1250) — matrix operand order repaired and checked by reading, not compiled, not run on hardware
//
#include <hip/hip_runtime.h>
#include <hip/hip_bf16.h>
#include <math.h>

// ---------------------------------------------------------------------------
// GIN GNN on MI455X (gfx1250, wave32).
// GEMMs: v_wmma_f32_16x16x32_bf16 with bf16x3 split (near-fp32 accuracy).
// All WMMA fragments are loaded as single contiguous 32-byte LDS reads
// (2x ds_load_b128) via fragment-ordered layouts:
//   B: weights pre-permuted in global memory: Wp[((k/32)*2 + (k/8)%2)*N + n][k%8 + 8*(k%32/16)]
//   A: activations stored with K-groups-of-8 permuted by g -> swap(g bit0,bit1)
//      within each group of 4, making each A fragment contiguous.
// Edge aggregation: one wave per edge, float4 per lane, agent-scope relaxed
// float atomics (global_atomic_add_f32).
// ---------------------------------------------------------------------------

typedef __attribute__((ext_vector_type(4)))  __bf16 v4bf;
typedef __attribute__((ext_vector_type(16))) __bf16 v16bf;
typedef __attribute__((ext_vector_type(8)))  float  v8f;

#define WMMA_BF16(a, b, c) \
  __builtin_amdgcn_wmma_f32_16x16x32_bf16(false, (a), false, (b), (short)0, (c), false, false)

static __device__ __forceinline__ void split_bf16(float x, __bf16& hi, __bf16& lo) {
  hi = (__bf16)x;
  lo = (__bf16)(x - (float)hi);
}

// K-group permutation (involution): within each block of 4 groups-of-8,
// swap groups 1 and 2.  perm: 0->0, 1->2, 2->1, 3->3.
static __device__ __forceinline__ int permg(int g) {
  return (g & ~3) | ((g & 1) << 1) | ((g >> 1) & 1);
}
// Permuted column position for activation storage.
static __device__ __forceinline__ int permcol(int col) {
  return permg(col >> 3) * 8 + (col & 7);
}

static __device__ __forceinline__ void split4(float x0, float x1, float x2, float x3,
                                              v4bf& hi, v4bf& lo) {
  __bf16 h0, l0, h1, l1, h2, l2, h3, l3;
  split_bf16(x0, h0, l0);
  split_bf16(x1, h1, l1);
  split_bf16(x2, h2, l2);
  split_bf16(x3, h3, l3);
  hi = (v4bf){h0, h1, h2, h3};
  lo = (v4bf){l0, l1, l2, l3};
}

// -------------------------------- utility ----------------------------------

__global__ void fill_zero_kernel(float* __restrict__ p, int n) {
  int i = blockIdx.x * blockDim.x + threadIdx.x;
  if (i < n) p[i] = 0.0f;
}

// Split weights into bf16 hi/lo, written in WMMA-B fragment order:
//   pos = (((k/32)*2 + (k%32/8)%2)*N + n)*16 + (k%8) + 8*(k%32/16)   for source element w[k*N + n].
__global__ void convert_split_perm_kernel(const float* __restrict__ w,
                                          __bf16* __restrict__ hi,
                                          __bf16* __restrict__ lo, int K, int N) {
  int i = blockIdx.x * blockDim.x + threadIdx.x;
  if (i < K * N) {
    int k = i / N, n = i - k * N;
    int kk = k & 31;
    int pos = (((k >> 5) * 2 + ((kk >> 3) & 1)) * N + n) * 16 + (kk & 7) + ((kk >> 4) << 3);
    __bf16 h, l;
    split_bf16(w[i], h, l);
    hi[pos] = h;
    lo[pos] = l;
  }
}

// ---------------------------- edge scatter-add ------------------------------
__global__ void scatter_add_kernel(const float* __restrict__ feat,
                                   const int* __restrict__ src,
                                   const int* __restrict__ dst,
                                   float* __restrict__ agg, int n_edges) {
  int e    = (blockIdx.x * blockDim.x + threadIdx.x) >> 5;
  int lane = threadIdx.x & 31;
  if (e >= n_edges) return;
  int s = src[e];
  int d = dst[e];
  const float4* v4 = (const float4*)(feat + (size_t)s * 128);
  float4 v = v4[lane];
  float* out = agg + (size_t)d * 128 + lane * 4;
  __hip_atomic_fetch_add(out + 0, v.x, __ATOMIC_RELAXED, __HIP_MEMORY_SCOPE_AGENT);
  __hip_atomic_fetch_add(out + 1, v.y, __ATOMIC_RELAXED, __HIP_MEMORY_SCOPE_AGENT);
  __hip_atomic_fetch_add(out + 2, v.z, __ATOMIC_RELAXED, __HIP_MEMORY_SCOPE_AGENT);
  __hip_atomic_fetch_add(out + 3, v.w, __ATOMIC_RELAXED, __HIP_MEMORY_SCOPE_AGENT);
}

// ------------------------- fragment load helpers ----------------------------
// A fragment (kc, half) = one contiguous v16bf at row*128 + kc*32 + half*16
// in the permuted activation layout.
static __device__ __forceinline__ void load_a_frags(const __bf16* __restrict__ sahi,
                                                    const __bf16* __restrict__ salo,
                                                    int arow, int half,
                                                    v16bf Ahi[4], v16bf Alo[4]) {
  const v16bf* ph = (const v16bf*)(sahi + arow * 128);
  const v16bf* pl = (const v16bf*)(salo + arow * 128);
#pragma unroll
  for (int kc = 0; kc < 4; ++kc) {
    Ahi[kc] = ph[kc * 2 + half];
    Alo[kc] = pl[kc * 2 + half];
  }
}

// ------------------------------- GIN MLP ------------------------------------
__launch_bounds__(256)
__global__ void gin_mlp_kernel(const float* __restrict__ xin,
                               const float* __restrict__ agg,
                               const __bf16* __restrict__ w1hi,
                               const __bf16* __restrict__ w1lo,
                               const float* __restrict__ b1,
                               const __bf16* __restrict__ w2hi,
                               const __bf16* __restrict__ w2lo,
                               const float* __restrict__ b2,
                               float* __restrict__ out, int n_nodes) {
  union SmemA {
    struct { __bf16 hi[128 * 128]; __bf16 lo[128 * 128]; } a;
    float h[128 * 128];
  };
  __shared__ __attribute__((aligned(16))) SmemA  smA;
  __shared__ __attribute__((aligned(16))) __bf16 sWhi[128 * 128];
  __shared__ __attribute__((aligned(16))) __bf16 sWlo[128 * 128];
  __shared__ float sM[128];
  __shared__ float sLS[128];

  const int tid  = threadIdx.x;
  const int base = blockIdx.x * 128;

  // Stage A: h = x + agg, split to bf16 hi/lo, permuted-K layout, float4 I/O.
  for (int i4 = tid; i4 < 128 * 32; i4 += 256) {
    int row = i4 >> 5, c4 = i4 & 31;
    int g = base + row;
    float x0 = 0.f, x1 = 0.f, x2 = 0.f, x3 = 0.f;
    if (g < n_nodes) {
      float4 a = ((const float4*)(xin + (size_t)g * 128))[c4];
      float4 b = ((const float4*)(agg + (size_t)g * 128))[c4];
      x0 = a.x + b.x; x1 = a.y + b.y; x2 = a.z + b.z; x3 = a.w + b.w;
    }
    v4bf hi, lo;
    split4(x0, x1, x2, x3, hi, lo);
    int pc = permcol(c4 * 4);  // 4-aligned within a group of 8 -> contiguous
    *(v4bf*)&smA.a.hi[row * 128 + pc] = hi;
    *(v4bf*)&smA.a.lo[row * 128 + pc] = lo;
  }
  // Load W1 (already fragment-ordered) into LDS with uint4 copies.
  {
    const uint4* ghi = (const uint4*)w1hi;
    const uint4* glo = (const uint4*)w1lo;
    uint4* shi = (uint4*)sWhi;
    uint4* slo = (uint4*)sWlo;
    for (int i = tid; i < 128 * 128 * 2 / 16; i += 256) {
      shi[i] = ghi[i];
      slo[i] = glo[i];
    }
  }
  __syncthreads();

  const int wv   = tid >> 5;
  const int lane = tid & 31;
  const int half = lane >> 4;
  const int lidx = lane & 15;
  const int arow = wv * 16 + lidx;

  v16bf Ahi[4], Alo[4];
  load_a_frags(smA.a.hi, smA.a.lo, arow, half, Ahi, Alo);

  const v16bf* pWh = (const v16bf*)sWhi;
  const v16bf* pWl = (const v16bf*)sWlo;

  // GEMM1: h1 = relu(h @ W1 + b1); split results back to own LDS rows.
#pragma unroll
  for (int nt = 0; nt < 8; ++nt) {
    v8f acc = {};
    const int bcol = nt * 16 + lidx;
#pragma unroll
    for (int kc = 0; kc < 4; ++kc) {
      v16bf Bhi = pWh[(kc * 2 + half) * 128 + bcol];
      v16bf Blo = pWl[(kc * 2 + half) * 128 + bcol];
      acc = WMMA_BF16(Ahi[kc], Bhi, acc);
      acc = WMMA_BF16(Ahi[kc], Blo, acc);
      acc = WMMA_BF16(Alo[kc], Bhi, acc);
    }
    const float bias = b1[bcol];
    const int pc = permcol(bcol);
#pragma unroll
    for (int v = 0; v < 8; ++v) {
      int row = wv * 16 + v + half * 8;
      float val = fmaxf(acc[v] + bias, 0.0f);
      __bf16 hi, lo;
      split_bf16(val, hi, lo);
      smA.a.hi[row * 128 + pc] = hi;
      smA.a.lo[row * 128 + pc] = lo;
    }
  }

  // Hoist h1 fragments before the union gets clobbered by other waves.
  load_a_frags(smA.a.hi, smA.a.lo, arow, half, Ahi, Alo);
  __syncthreads();

  // Swap W2 into LDS.
  {
    const uint4* ghi = (const uint4*)w2hi;
    const uint4* glo = (const uint4*)w2lo;
    uint4* shi = (uint4*)sWhi;
    uint4* slo = (uint4*)sWlo;
    for (int i = tid; i < 128 * 128 * 2 / 16; i += 256) {
      shi[i] = ghi[i];
      slo[i] = glo[i];
    }
  }
  __syncthreads();

  // GEMM2: h2 = h1 @ W2 + b2 ; f32 results into LDS (reuses activation union).
#pragma unroll
  for (int nt = 0; nt < 8; ++nt) {
    v8f acc = {};
    const int bcol = nt * 16 + lidx;
#pragma unroll
    for (int kc = 0; kc < 4; ++kc) {
      v16bf Bhi = pWh[(kc * 2 + half) * 128 + bcol];
      v16bf Blo = pWl[(kc * 2 + half) * 128 + bcol];
      acc = WMMA_BF16(Ahi[kc], Bhi, acc);
      acc = WMMA_BF16(Ahi[kc], Blo, acc);
      acc = WMMA_BF16(Alo[kc], Bhi, acc);
    }
    const float bias = b2[bcol];
#pragma unroll
    for (int v = 0; v < 8; ++v) {
      int row = wv * 16 + v + half * 8;
      smA.h[row * 128 + bcol] = acc[v] + bias;
    }
  }
  __syncthreads();

  // Row-wise log_softmax parameters (one thread per row), float4 LDS reads.
  if (tid < 128) {
    const float4* ph = (const float4*)&smA.h[tid * 128];
    float m = -1e30f;
    for (int c = 0; c < 32; ++c) {
      float4 v = ph[c];
      m = fmaxf(m, fmaxf(fmaxf(v.x, v.y), fmaxf(v.z, v.w)));
    }
    float s = 0.0f;
    for (int c = 0; c < 32; ++c) {
      float4 v = ph[c];
      s += expf(v.x - m) + expf(v.y - m) + expf(v.z - m) + expf(v.w - m);
    }
    sM[tid]  = m;
    sLS[tid] = logf(s);
  }
  __syncthreads();

  // Cooperative, coalesced store of log_softmax output.
  for (int i4 = tid; i4 < 128 * 32; i4 += 256) {
    int row = i4 >> 5, c4 = i4 & 31;
    int g = base + row;
    if (g < n_nodes) {
      float4 v = ((const float4*)&smA.h[row * 128])[c4];
      float sh = sM[row] + sLS[row];
      float4 o;
      o.x = v.x - sh; o.y = v.y - sh; o.z = v.z - sh; o.w = v.w - sh;
      ((float4*)(out + (size_t)g * 128))[c4] = o;
    }
  }
}

// ------------------------------ output head ---------------------------------
__launch_bounds__(256)
__global__ void head_kernel(const float* __restrict__ xin,
                            const __bf16* __restrict__ whi_g,
                            const __bf16* __restrict__ wlo_g,
                            const float* __restrict__ bo,
                            float* __restrict__ out, int n_nodes) {
  union SmemA {
    struct { __bf16 hi[128 * 128]; __bf16 lo[128 * 128]; } a;
    float h[128 * 64];
  };
  __shared__ __attribute__((aligned(16))) SmemA  smA;
  __shared__ __attribute__((aligned(16))) __bf16 sWhi[128 * 64];
  __shared__ __attribute__((aligned(16))) __bf16 sWlo[128 * 64];
  __shared__ float sM[128];
  __shared__ float sInv[128];

  const int tid  = threadIdx.x;
  const int base = blockIdx.x * 128;

  for (int i4 = tid; i4 < 128 * 32; i4 += 256) {
    int row = i4 >> 5, c4 = i4 & 31;
    int g = base + row;
    float x0 = 0.f, x1 = 0.f, x2 = 0.f, x3 = 0.f;
    if (g < n_nodes) {
      float4 a = ((const float4*)(xin + (size_t)g * 128))[c4];
      x0 = a.x; x1 = a.y; x2 = a.z; x3 = a.w;
    }
    v4bf hi, lo;
    split4(x0, x1, x2, x3, hi, lo);
    int pc = permcol(c4 * 4);
    *(v4bf*)&smA.a.hi[row * 128 + pc] = hi;
    *(v4bf*)&smA.a.lo[row * 128 + pc] = lo;
  }
  {
    const uint4* ghi = (const uint4*)whi_g;
    const uint4* glo = (const uint4*)wlo_g;
    uint4* shi = (uint4*)sWhi;
    uint4* slo = (uint4*)sWlo;
    for (int i = tid; i < 128 * 64 * 2 / 16; i += 256) {
      shi[i] = ghi[i];
      slo[i] = glo[i];
    }
  }
  __syncthreads();

  const int wv   = tid >> 5;
  const int lane = tid & 31;
  const int half = lane >> 4;
  const int lidx = lane & 15;
  const int arow = wv * 16 + lidx;

  v16bf Ahi[4], Alo[4];
  load_a_frags(smA.a.hi, smA.a.lo, arow, half, Ahi, Alo);

  const v16bf* pWh = (const v16bf*)sWhi;
  const v16bf* pWl = (const v16bf*)sWlo;

#pragma unroll
  for (int nt = 0; nt < 4; ++nt) {
    v8f acc = {};
    const int bcol = nt * 16 + lidx;
#pragma unroll
    for (int kc = 0; kc < 4; ++kc) {
      v16bf Bhi = pWh[(kc * 2 + half) * 64 + bcol];
      v16bf Blo = pWl[(kc * 2 + half) * 64 + bcol];
      acc = WMMA_BF16(Ahi[kc], Bhi, acc);
      acc = WMMA_BF16(Ahi[kc], Blo, acc);
      acc = WMMA_BF16(Alo[kc], Bhi, acc);
    }
    const float bias = bo[bcol];
#pragma unroll
    for (int v = 0; v < 8; ++v) {
      int row = wv * 16 + v + half * 8;
      smA.h[row * 64 + bcol] = acc[v] + bias;   // clobbers only own rows of a.hi
    }
  }
  __syncthreads();

  if (tid < 128) {
    const float4* ph = (const float4*)&smA.h[tid * 64];
    float m = -1e30f;
    for (int c = 0; c < 16; ++c) {
      float4 v = ph[c];
      m = fmaxf(m, fmaxf(fmaxf(v.x, v.y), fmaxf(v.z, v.w)));
    }
    float s = 0.0f;
    for (int c = 0; c < 16; ++c) {
      float4 v = ph[c];
      s += expf(v.x - m) + expf(v.y - m) + expf(v.z - m) + expf(v.w - m);
    }
    sM[tid]   = m;
    sInv[tid] = 1.0f / s;
  }
  __syncthreads();

  for (int i4 = tid; i4 < 128 * 16; i4 += 256) {
    int row = i4 >> 4, c4 = i4 & 15;
    int g = base + row;
    if (g < n_nodes) {
      float4 v = ((const float4*)&smA.h[row * 64])[c4];
      float m = sM[row], inv = sInv[row];
      float4 o;
      o.x = expf(v.x - m) * inv; o.y = expf(v.y - m) * inv;
      o.z = expf(v.z - m) * inv; o.w = expf(v.w - m) * inv;
      ((float4*)(out + (size_t)g * 64))[c4] = o;
    }
  }
}

// ------------------------------- launcher -----------------------------------

extern "C" void kernel_launch(void* const* d_in, const int* in_sizes, int n_in,
                              void* d_out, int out_size, void* d_ws, size_t ws_size,
                              hipStream_t stream) {
  const float* x    = (const float*)d_in[0];
  const int*   ei   = (const int*)d_in[1];
  const float* w1_0 = (const float*)d_in[2];
  const float* b1_0 = (const float*)d_in[3];
  const float* w2_0 = (const float*)d_in[4];
  const float* b2_0 = (const float*)d_in[5];
  const float* w1_1 = (const float*)d_in[6];
  const float* b1_1 = (const float*)d_in[7];
  const float* w2_1 = (const float*)d_in[8];
  const float* b2_1 = (const float*)d_in[9];
  const float* wo   = (const float*)d_in[10];
  const float* bo   = (const float*)d_in[11];

  const int n_nodes = in_sizes[0] / 128;
  const int n_edges = in_sizes[1] / 2;
  const int* src = ei;
  const int* dst = ei + n_edges;

  char* ws = (char*)d_ws;
  size_t off = 0;
  float* agg  = (float*)(ws + off); off += (size_t)n_nodes * 128 * sizeof(float);
  float* xbuf = (float*)(ws + off); off += (size_t)n_nodes * 128 * sizeof(float);
  const int WN = 128 * 128, WO = 128 * 64;
  __bf16* w1_0hi = (__bf16*)(ws + off); off += WN * 2;
  __bf16* w1_0lo = (__bf16*)(ws + off); off += WN * 2;
  __bf16* w2_0hi = (__bf16*)(ws + off); off += WN * 2;
  __bf16* w2_0lo = (__bf16*)(ws + off); off += WN * 2;
  __bf16* w1_1hi = (__bf16*)(ws + off); off += WN * 2;
  __bf16* w1_1lo = (__bf16*)(ws + off); off += WN * 2;
  __bf16* w2_1hi = (__bf16*)(ws + off); off += WN * 2;
  __bf16* w2_1lo = (__bf16*)(ws + off); off += WN * 2;
  __bf16* wohi   = (__bf16*)(ws + off); off += WO * 2;
  __bf16* wolo   = (__bf16*)(ws + off); off += WO * 2;

  // Pre-split weights into bf16 hi/lo in WMMA fragment order.
  convert_split_perm_kernel<<<(WN + 255) / 256, 256, 0, stream>>>(w1_0, w1_0hi, w1_0lo, 128, 128);
  convert_split_perm_kernel<<<(WN + 255) / 256, 256, 0, stream>>>(w2_0, w2_0hi, w2_0lo, 128, 128);
  convert_split_perm_kernel<<<(WN + 255) / 256, 256, 0, stream>>>(w1_1, w1_1hi, w1_1lo, 128, 128);
  convert_split_perm_kernel<<<(WN + 255) / 256, 256, 0, stream>>>(w2_1, w2_1hi, w2_1lo, 128, 128);
  convert_split_perm_kernel<<<(WO + 255) / 256, 256, 0, stream>>>(wo, wohi, wolo, 128, 64);

  const int feat_n    = n_nodes * 128;
  const int fill_grid = (feat_n + 255) / 256;
  const int scat_grid = (n_edges + 7) / 8;   // 8 waves (edges) per block
  const int mlp_grid  = (n_nodes + 127) / 128;

  // ---- layer 0 ----
  fill_zero_kernel<<<fill_grid, 256, 0, stream>>>(agg, feat_n);
  scatter_add_kernel<<<scat_grid, 256, 0, stream>>>(x, src, dst, agg, n_edges);
  gin_mlp_kernel<<<mlp_grid, 256, 0, stream>>>(x, agg, w1_0hi, w1_0lo, b1_0,
                                               w2_0hi, w2_0lo, b2_0, xbuf, n_nodes);
  // ---- layer 1 (xbuf updated in place; each block touches only its rows) ----
  fill_zero_kernel<<<fill_grid, 256, 0, stream>>>(agg, feat_n);
  scatter_add_kernel<<<scat_grid, 256, 0, stream>>>(xbuf, src, dst, agg, n_edges);
  gin_mlp_kernel<<<mlp_grid, 256, 0, stream>>>(xbuf, agg, w1_1hi, w1_1lo, b1_1,
                                               w2_1hi, w2_1lo, b2_1, xbuf, n_nodes);
  // ---- head ----
  head_kernel<<<mlp_grid, 256, 0, stream>>>(xbuf, wohi, wolo, bo,
                                            (float*)d_out, n_nodes);
}
